// Clustering_64604898066647
// MI455X (gfx1250) — matrix operand order repaired and checked by reading, not compiled, not run on hardware
//
#include <hip/hip_runtime.h>
#include <hip/hip_bf16.h>

typedef __attribute__((ext_vector_type(16))) _Float16 v16h;
typedef __attribute__((ext_vector_type(8)))  float    v8f;

constexpr int NB  = 16;    // batch
constexpr int NH  = 16;    // heads
constexpr int NL  = 2048;  // query length
constexpr int ND  = 64;    // head dim
constexpr int NLK = 4096;  // key length
constexpr int NC  = 10;    // clusters
constexpr int NS  = 16;    // shrunk key length (LKS)

// ---------------- Pass 1: shrink K/V along key-sequence dim ----------------
// grid = 512 (bh*2 + mat), block = 256 (4 k-groups x 64 d). Streams K/V once
// (coalesced b32), weight rows are block-uniform -> scalar loads. Deterministic
// 4-way LDS reduction over k-quarters.
__global__ __launch_bounds__(256) void shrink_kernel(
    const float* __restrict__ K, const float* __restrict__ V,
    const float* __restrict__ kw, const float* __restrict__ kb,
    const float* __restrict__ vw, const float* __restrict__ vb,
    float* __restrict__ Ks, float* __restrict__ Vs)
{
  __shared__ float red[4][NS][ND];
  int bh  = blockIdx.x >> 1;
  int mat = blockIdx.x & 1;
  int tid = threadIdx.x;
  int kg  = tid >> 6;
  int d   = tid & 63;
  const float* src = mat ? V : K;
  const float* w   = mat ? vw : kw;
  const float* bia = mat ? vb : kb;
  float*       dst = mat ? Vs : Ks;
  const float* col = src + (size_t)bh * NLK * ND + d;

  float acc[NS];
#pragma unroll
  for (int s = 0; s < NS; ++s) acc[s] = 0.f;

  int k0 = kg * (NLK / 4);
  for (int k = k0; k < k0 + NLK / 4; k += 4) {
    float v0 = col[(k + 0) * ND];
    float v1 = col[(k + 1) * ND];
    float v2 = col[(k + 2) * ND];
    float v3 = col[(k + 3) * ND];
#pragma unroll
    for (int s = 0; s < NS; ++s) {
      const float* wr = w + s * NLK + k;   // block-uniform -> s_load_b128
      acc[s] += wr[0] * v0 + wr[1] * v1 + wr[2] * v2 + wr[3] * v3;
    }
  }
#pragma unroll
  for (int s = 0; s < NS; ++s) red[kg][s][d] = acc[s];
  __syncthreads();
  if (kg == 0) {
#pragma unroll
    for (int s = 0; s < NS; ++s) {
      float v = red[0][s][d] + red[1][s][d] + red[2][s][d] + red[3][s][d] + bia[s];
      dst[(bh * NS + s) * ND + d] = v;
    }
  }
}

// ---------------- Pass 2: cluster pipeline + loss partials ----------------
__device__ inline float softplusf(float x) {
  return (x > 20.f) ? x : log1pf(expf(x));
}

// grid = 256 (b*16+s), block = 256; each thread handles 4 (h,d) positions.
// cluster_q kept in LDS so the CE log-softmax across h stays in-block.
__global__ __launch_bounds__(256) void cluster_kernel(
    const float* __restrict__ Ks,
    const float* __restrict__ pw,  const float* __restrict__ pb,
    const float* __restrict__ ckw, const float* __restrict__ ckb,
    const float* __restrict__ cqw, const float* __restrict__ cqb,
    const float* __restrict__ muw, const float* __restrict__ mub,
    const float* __restrict__ sgw, const float* __restrict__ sgb,
    unsigned char* __restrict__ cls,
    float* __restrict__ lp_part, float* __restrict__ ce_part)
{
  __shared__ float W[640];
  __shared__ float qlds[NH * ND * NC];   // 40 KB: cluster_q[h][d][c]
  __shared__ float red[512];

  int bid = blockIdx.x;
  int b = bid >> 4, s = bid & 15;
  int tid = threadIdx.x;

  if (tid < 160) W[tid] = pw[tid];
  if (tid < 100) {
    W[160 + tid] = ckw[tid]; W[260 + tid] = cqw[tid];
    W[360 + tid] = muw[tid]; W[460 + tid] = sgw[tid];
  }
  if (tid < 10) {
    W[560 + tid] = pb[tid];  W[570 + tid] = ckb[tid]; W[580 + tid] = cqb[tid];
    W[590 + tid] = mub[tid]; W[600 + tid] = sgb[tid];
  }
  __syncthreads();

  float lp = 0.f;
  for (int pi = 0; pi < 4; ++pi) {
    int p = tid + pi * 256;
    int h = p >> 6, d = p & 63;

    // unfold window over batch dim: K_padded[b+j] = (b+j>=16) ? Ks[b+j-15] : 0
    float ku[16];
#pragma unroll
    for (int j = 0; j < 16; ++j) {
      int bj = b + j;
      ku[j] = (bj >= 16) ? Ks[(((bj - 15) * NH + h) * NS + s) * ND + d] : 0.f;
    }
    // proj_to_cluster_k + ReLU
    float pc[NC];
#pragma unroll
    for (int c = 0; c < NC; ++c) {
      float v = W[560 + c];
#pragma unroll
      for (int j = 0; j < 16; ++j) v += W[c * 16 + j] * ku[j];
      pc[c] = fmaxf(v, 0.f);
    }
    float qk[NC], qq[NC];
    { // cluster_k = softmax(ck)
      float mx = -3.4e38f;
#pragma unroll
      for (int c = 0; c < NC; ++c) {
        float t = W[570 + c];
#pragma unroll
        for (int c2 = 0; c2 < NC; ++c2) t += W[160 + c * 10 + c2] * pc[c2];
        qk[c] = t; mx = fmaxf(mx, t);
      }
      float sm = 0.f;
#pragma unroll
      for (int c = 0; c < NC; ++c) { float e = expf(qk[c] - mx); qk[c] = e; sm += e; }
      float inv = 1.f / sm;
#pragma unroll
      for (int c = 0; c < NC; ++c) qk[c] *= inv;
    }
    { // cluster_q = softmax(cq)
      float mx = -3.4e38f;
#pragma unroll
      for (int c = 0; c < NC; ++c) {
        float t = W[580 + c];
#pragma unroll
        for (int c2 = 0; c2 < NC; ++c2) t += W[260 + c * 10 + c2] * pc[c2];
        qq[c] = t; mx = fmaxf(mx, t);
      }
      float sm = 0.f;
#pragma unroll
      for (int c = 0; c < NC; ++c) { float e = expf(qq[c] - mx); qq[c] = e; sm += e; }
      float inv = 1.f / sm;
#pragma unroll
      for (int c = 0; c < NC; ++c) qq[c] *= inv;
    }
    // mu, sigma, Normal log_prob of cluster_k
#pragma unroll
    for (int c = 0; c < NC; ++c) {
      float mu = W[590 + c], sa = W[600 + c];
#pragma unroll
      for (int c2 = 0; c2 < NC; ++c2) {
        mu += W[360 + c * 10 + c2] * qq[c2];
        sa += W[460 + c * 10 + c2] * qq[c2];
      }
      float sg = softplusf(sa);
      float z = (qk[c] - mu) / sg;
      lp += -0.5f * z * z - logf(sg) - 0.9189385332046727f; // 0.5*log(2*pi)
    }
    // argmax (first max wins, like jnp.argmax)
    int am = 0; float bv = qq[0];
#pragma unroll
    for (int c = 1; c < NC; ++c) if (qq[c] > bv) { bv = qq[c]; am = c; }
    cls[((b * NH + h) * NS + s) * ND + d] = (unsigned char)am;
#pragma unroll
    for (int c = 0; c < NC; ++c) qlds[p * NC + c] = qq[c];
  }
  __syncthreads();

  // CE: log_softmax over h for each (d,c); sum q*(q-lse) over h
  float ce = 0.f;
  for (int combo = tid; combo < ND * NC; combo += 256) {
    int d = combo / NC, c = combo % NC;
    float q[NH]; float mx = -3.4e38f;
#pragma unroll
    for (int h = 0; h < NH; ++h) { q[h] = qlds[((h << 6) + d) * NC + c]; mx = fmaxf(mx, q[h]); }
    float sm = 0.f;
#pragma unroll
    for (int h = 0; h < NH; ++h) sm += expf(q[h] - mx);
    float lse = mx + logf(sm);
#pragma unroll
    for (int h = 0; h < NH; ++h) ce += q[h] * (q[h] - lse);
  }

  red[tid] = lp; red[256 + tid] = ce;
  __syncthreads();
  for (int off = 128; off > 0; off >>= 1) {
    if (tid < off) { red[tid] += red[tid + off]; red[256 + tid] += red[256 + tid + off]; }
    __syncthreads();
  }
  if (tid == 0) { lp_part[bid] = red[0]; ce_part[bid] = red[256]; }
}

// ---------------- deterministic loss finalize ----------------
__global__ void loss_kernel(const float* __restrict__ lp_part,
                            const float* __restrict__ ce_part,
                            float* __restrict__ out_loss)
{
  if (threadIdx.x == 0) {
    float lp = 0.f, ce = 0.f;
    for (int i = 0; i < 256; ++i) { lp += lp_part[i]; ce += ce_part[i]; }
    // loss = -mean(log_prob) + (-mean_{b,s,d,c} sum_h q*logp)
    *out_loss = -(lp / 2621440.0f) - (ce / 163840.0f);
  }
}

// ---------------- Pass 3: scores (WMMA f16 hi/lo) + softmax + context -------
// grid = 256 (bh), block = 256 (8 waves). S = Q x Ind^T with column j = c*16+k,
// so tile t == cluster c and max-over-c is a register fmax across tiles.
// B-fragments (0/1 indicator, exact in f16) are hoisted to registers by the
// compiler and reused across all 16 row-block iterations.
__global__ __launch_bounds__(256) void attn_kernel(
    const float* __restrict__ Q, const float* __restrict__ Vs,
    const unsigned char* __restrict__ cls, float* __restrict__ out)
{
  __shared__ __align__(32) _Float16 bfrag[NC * 2 * 32 * 16]; // [c][khalf][lane][i]
  __shared__ float vs_l[NS * ND];
  __shared__ float aw_l[8][NS * NS];

  int bh   = blockIdx.x;
  int tid  = threadIdx.x;
  int lane = tid & 31, wave = tid >> 5;

  // Build B fragments: Ind^T[(d),(c*16+k)] = (cls[k][d]==c), 0/1 exact in f16.
  // B layout (32x16, K=d): lane -> N=k (lane&15); half i -> K = (i<8?i:i+8) + 8*(lane>=16).
  const unsigned char* clsb = cls + bh * (NS * ND);
  for (int idx = tid; idx < NC * 2 * 32 * 16; idx += 256) {
    int i  = idx & 15;
    int ln = (idx >> 4) & 31;
    int kh = (idx >> 9) & 1;
    int c  = idx >> 10;
    int k  = ln & 15;
    int dK = kh * 32 + ((i < 8) ? i : i + 8) + ((ln < 16) ? 0 : 8);
    bfrag[idx] = (clsb[k * ND + dK] == c) ? (_Float16)1.0f : (_Float16)0.0f;
  }
  for (int idx = tid; idx < NS * ND; idx += 256) vs_l[idx] = Vs[bh * (NS * ND) + idx];
  __syncthreads();

  const float* Qb = Q   + (size_t)bh * NL * ND;
  float*       Ob = out + (size_t)bh * NL * ND;
  float* aw = aw_l[wave];
  int m    = lane & 15;
  int koff = (lane < 16) ? 0 : 8;

  // hoist Vs columns for this lane; lane owns d = 2*lane, 2*lane+1 so the
  // context stores become one coalesced b64 per lane per row.
  float vcolA[NS], vcolB[NS];
#pragma unroll
  for (int k = 0; k < NS; ++k) {
    vcolA[k] = vs_l[k * ND + 2 * lane];
    vcolB[k] = vs_l[k * ND + 2 * lane + 1];
  }

  for (int blk = wave; blk < NL / 16; blk += 8) {
    int qbase = blk * 16;
    const float* row = Qb + (size_t)(qbase + m) * ND;

    // A fragments (16x32 f16): lane m=row; half i -> K = (i<8?i:i+8)+8*(lane>=16).
    // hi/lo split of Q for fp32-accurate accumulation.
    v16h a0h, a0l, a1h, a1l;
#pragma unroll
    for (int i2 = 0; i2 < 8; ++i2) {
      int kp = ((i2 < 4) ? 2 * i2 : 2 * i2 + 8) + koff;
      float x0 = row[kp], x1 = row[kp + 1];
      _Float16 h0 = (_Float16)x0, h1 = (_Float16)x1;
      a0h[2 * i2] = h0;  a0h[2 * i2 + 1] = h1;
      a0l[2 * i2]     = (_Float16)(x0 - (float)h0);
      a0l[2 * i2 + 1] = (_Float16)(x1 - (float)h1);
      float y0 = row[32 + kp], y1 = row[32 + kp + 1];
      _Float16 g0 = (_Float16)y0, g1 = (_Float16)y1;
      a1h[2 * i2] = g0;  a1h[2 * i2 + 1] = g1;
      a1l[2 * i2]     = (_Float16)(y0 - (float)g0);
      a1l[2 * i2 + 1] = (_Float16)(y1 - (float)g1);
    }

    // prefetch next Q tile (both 128B lines of each of this lane's rows)
    if (blk + 8 < NL / 16) {
      const float* nrow = Qb + (size_t)(qbase + 128 + m) * ND;
      __builtin_prefetch(nrow + koff, 0, 1);
      __builtin_prefetch(nrow + 32 + koff, 0, 1);
    }

    float best[8];
#pragma unroll
    for (int r = 0; r < 8; ++r) best[r] = -3.4e38f;

#pragma unroll
    for (int c = 0; c < NC; ++c) {
      const v16h b0 = *(const v16h*)&bfrag[(c * 2 + 0) * 512 + lane * 16];
      const v16h b1 = *(const v16h*)&bfrag[(c * 2 + 1) * 512 + lane * 16];
      v8f acc = {};
      acc = __builtin_amdgcn_wmma_f32_16x16x32_f16(false, a0h, false, b0, (short)0, acc, false, false);
      acc = __builtin_amdgcn_wmma_f32_16x16x32_f16(false, a1h, false, b1, (short)0, acc, false, false);
      acc = __builtin_amdgcn_wmma_f32_16x16x32_f16(false, a0l, false, b0, (short)0, acc, false, false);
      acc = __builtin_amdgcn_wmma_f32_16x16x32_f16(false, a1l, false, b1, (short)0, acc, false, false);
#pragma unroll
      for (int r = 0; r < 8; ++r) best[r] = fmaxf(best[r], acc[r]);
    }

    // final_score[q][k] = 0.1 * max_c S ; C-layout: row = r + koff, col = m
#pragma unroll
    for (int r = 0; r < 8; ++r) aw[(r + koff) * NS + m] = 0.1f * best[r];

    // softmax over k (one q row per lane 0..15); wave-private LDS, DS in-order
    if (lane < 16) {
      float* rowa = aw + lane * NS;
      float mx = rowa[0];
#pragma unroll
      for (int k = 1; k < NS; ++k) mx = fmaxf(mx, rowa[k]);
      float sm = 0.f;
#pragma unroll
      for (int k = 0; k < NS; ++k) { float e = expf(rowa[k] - mx); rowa[k] = e; sm += e; }
      float inv = 1.f / sm;
#pragma unroll
      for (int k = 0; k < NS; ++k) rowa[k] *= inv;
    }

    // context = attn @ Vs (exact fp32 VALU), one b64 store per lane per row
    for (int qi = 0; qi < 16; ++qi) {
      float a0 = 0.f, a1 = 0.f;
#pragma unroll
      for (int k = 0; k < NS; ++k) {
        float at = aw[qi * NS + k];
        a0 += at * vcolA[k];
        a1 += at * vcolB[k];
      }
      float2 st; st.x = a0; st.y = a1;
      *(float2*)&Ob[(size_t)(qbase + qi) * ND + 2 * lane] = st;
    }
  }
}

extern "C" void kernel_launch(void* const* d_in, const int* in_sizes, int n_in,
                              void* d_out, int out_size, void* d_ws, size_t ws_size,
                              hipStream_t stream) {
  (void)in_sizes; (void)n_in; (void)out_size; (void)ws_size;
  const float* Q   = (const float*)d_in[0];
  const float* K   = (const float*)d_in[1];
  const float* V   = (const float*)d_in[2];
  const float* skw = (const float*)d_in[3];
  const float* skb = (const float*)d_in[4];
  const float* svw = (const float*)d_in[5];
  const float* svb = (const float*)d_in[6];
  const float* pw  = (const float*)d_in[7];
  const float* pb  = (const float*)d_in[8];
  const float* ckw = (const float*)d_in[9];
  const float* ckb = (const float*)d_in[10];
  const float* cqw = (const float*)d_in[11];
  const float* cqb = (const float*)d_in[12];
  const float* muw = (const float*)d_in[13];
  const float* mub = (const float*)d_in[14];
  const float* sgw = (const float*)d_in[15];
  const float* sgb = (const float*)d_in[16];
  float* out = (float*)d_out;

  // workspace layout (no zero-init needed: everything fully written each call)
  char* ws = (char*)d_ws;
  float* Ks = (float*)ws;                               // 262144 f32
  float* Vs = Ks + NB * NH * NS * ND;                   // 262144 f32
  unsigned char* cls = (unsigned char*)(Vs + NB * NH * NS * ND); // 262144 u8
  float* lp_part = (float*)(cls + NB * NH * NS * ND);   // 256 f32
  float* ce_part = lp_part + 256;                       // 256 f32

  shrink_kernel <<<NB * NH * 2, 256, 0, stream>>>(K, V, skw, skb, svw, svb, Ks, Vs);
  cluster_kernel<<<NB * NS,     256, 0, stream>>>(Ks, pw, pb, ckw, ckb, cqw, cqb,
                                                  muw, mub, sgw, sgb, cls, lp_part, ce_part);
  loss_kernel   <<<1,            32, 0, stream>>>(lp_part, ce_part, out + (size_t)NB * NH * NL * ND);
  attn_kernel   <<<NB * NH,     256, 0, stream>>>(Q, Vs, cls, out);
}
